// BertBiLSTM_CRF_90202903151342
// MI455X (gfx1250) — compile-verified, matrix operand order checked by reading, not run on hardware
//
#include <hip/hip_runtime.h>

#define S_LEN 8192
#define EMB_D 768
#define HD_D  512
#define G4_D  2048     // 4 * HD
#define HID_D 1024
#define NT    14
#define START_TAG 12
#define STOP_TAG  13
#define NEG_INF  -10000.0f

typedef __attribute__((ext_vector_type(16))) __bf16 v16bf;
typedef __attribute__((ext_vector_type(8)))  float  v8f;

__device__ __forceinline__ unsigned short f2bf(float f) {
  unsigned int u = __builtin_bit_cast(unsigned int, f);
  u += 0x7FFFu + ((u >> 16) & 1u);           // round-to-nearest-even
  return (unsigned short)(u >> 16);
}
__device__ __forceinline__ float bf_lo(unsigned int u) {  // low u16 as bf16->f32
  return __builtin_bit_cast(float, u << 16);
}
__device__ __forceinline__ float sigmoidf_(float x) { return 1.0f / (1.0f + __expf(-x)); }

// CDNA5 async memory->LDS (tracked by ASYNCcnt, no VGPR round trip).
// dst: LDS byte address in a VGPR; src: 64-bit global address in a VGPR pair.
__device__ __forceinline__ void async_ld_b128(unsigned lds_off, const void* gptr) {
  asm volatile("global_load_async_to_lds_b128 %0, %1, off"
               :: "v"(lds_off), "v"((unsigned long long)(uintptr_t)gptr)
               : "memory");
}
__device__ __forceinline__ void wait_async0() {
  asm volatile("s_wait_asynccnt 0x0" ::: "memory");
}

// ---------------------------------------------------------------------------
// Conversion kernels
// ---------------------------------------------------------------------------
__global__ void k_f32_to_bf16(const float* __restrict__ src,
                              unsigned short* __restrict__ dst, int n) {
  int i = blockIdx.x * blockDim.x + threadIdx.x;
  if (i < n) dst[i] = f2bf(src[i]);
}

// w_hh [2048][512] f32  ->  wt [512][2048] bf16 (transposed for coalesced GEMV)
__global__ void k_whh_transpose(const float* __restrict__ whh,
                                unsigned short* __restrict__ wt) {
  int i = blockIdx.x * blockDim.x + threadIdx.x;
  if (i < G4_D * HD_D) {
    int g = i >> 9, k = i & (HD_D - 1);
    wt[k * G4_D + g] = f2bf(whh[i]);
  }
}

__global__ void k_init_sync(int* cnt) {
  if (threadIdx.x < 4) cnt[threadIdx.x] = 0;
}

// ---------------------------------------------------------------------------
// Input projection:  xg[dir][s][g] = bf16(x) @ bf16(W_ih)^T + b_ih + b_hh
// 128x128 block tile, BK=32. 8 waves, each wave: 4(M) x 2(N) 16x16 WMMA tiles.
// Double-buffered LDS; tiles staged with global_load_async_to_lds_b128 and
// overlapped with the WMMA work of the previous K-step (one barrier / step).
// ---------------------------------------------------------------------------
#define BM 128
#define BN 128
#define KSTEPS (EMB_D / 32)   // 24
#define TILE_U32 (BM * 16)    // 16 u32 K-pairs per row, 8KB per tile

__device__ __forceinline__ void stage_tile_async(
    const unsigned short* __restrict__ X, const unsigned short* __restrict__ W,
    int m0, int n0, int kb, unsigned aoff, unsigned boff, int tid) {
  // A tile: 128 rows x 32 K (bf16) = 512 x 16B chunks, 2 per thread
#pragma unroll
  for (int l = tid; l < 512; l += 256) {
    const int row = l >> 2, seg = l & 3;
    async_ld_b128(aoff + (unsigned)(row * 16 + seg * 4) * 4u,
                  X + (size_t)(m0 + row) * EMB_D + kb + seg * 8);
  }
  // B tile: 128 gate-cols x 32 K
#pragma unroll
  for (int l = tid; l < 512; l += 256) {
    const int row = l >> 2, seg = l & 3;
    async_ld_b128(boff + (unsigned)(row * 16 + seg * 4) * 4u,
                  W + (size_t)(n0 + row) * EMB_D + kb + seg * 8);
  }
}

__global__ __launch_bounds__(256)
void k_xproj_wmma(const unsigned short* __restrict__ Xb,   // [S][768] bf16
                  const unsigned short* __restrict__ Wf,   // [2048][768] bf16
                  const unsigned short* __restrict__ Wb,
                  const float* __restrict__ b_ih_f, const float* __restrict__ b_hh_f,
                  const float* __restrict__ b_ih_b, const float* __restrict__ b_hh_b,
                  float* __restrict__ xg) {                // [2][S][2048]
  const int dir = blockIdx.z;
  const unsigned short* W = dir ? Wb : Wf;
  const float* bi = dir ? b_ih_b : b_ih_f;
  const float* bh = dir ? b_hh_b : b_hh_f;
  float* out = xg + (size_t)dir * S_LEN * G4_D;

  const int n0 = blockIdx.x * BN;
  const int m0 = blockIdx.y * BM;
  const int tid  = threadIdx.x;
  const int lane = tid & 31;
  const int wv   = tid >> 5;      // 0..7
  const int wm   = wv >> 2;       // 0..1 -> 64 rows
  const int wn   = wv & 3;        // 0..3 -> 32 cols

  // no padding: row stride 64B keeps every 16B fragment chunk aligned for
  // async b128 writes and ds_load_b128 fragment reads
  __shared__ unsigned int As[2][TILE_U32];
  __shared__ unsigned int Bs[2][TILE_U32];

  const unsigned a_base = (unsigned)(uintptr_t)(&As[0][0]);
  const unsigned b_base = (unsigned)(uintptr_t)(&Bs[0][0]);

  v8f acc[4][2];
  for (int i = 0; i < 4; ++i)
    for (int j = 0; j < 2; ++j)
      for (int e = 0; e < 8; ++e) acc[i][j][e] = 0.0f;

  const int r16 = lane & 15;
  const int pb  = (lane >> 4) * 4;   // pair-base: 0 or 4 (K offset 0 / 8)

  // prologue: stage K-step 0 into buffer 0
  stage_tile_async(Xb, W, m0, n0, 0, a_base, b_base, tid);

  for (int it = 0; it < KSTEPS; ++it) {
    wait_async0();          // our async writes for buf_cur have landed
    __syncthreads();        // everyone's landed; everyone done reading buf_next

    const int buf = it & 1;
    if (it + 1 < KSTEPS)    // overlap next tile's async loads with this WMMA work
      stage_tile_async(Xb, W, m0, n0, (it + 1) * 32,
                       a_base + (unsigned)(buf ^ 1) * (TILE_U32 * 4u),
                       b_base + (unsigned)(buf ^ 1) * (TILE_U32 * 4u), tid);

    const unsigned int* Ab = &As[buf][0];
    const unsigned int* Bb = &Bs[buf][0];

    // ---- gather fragments per 16-bit A-matrix VGPR layout (2x ds_load_b128) ----
    v16bf afrag[4], bfrag[2];
#pragma unroll
    for (int mi = 0; mi < 4; ++mi) {
      const int row = wm * 64 + mi * 16 + r16;
      const uint4 c0 = *(const uint4*)(Ab + row * 16 + pb);
      const uint4 c1 = *(const uint4*)(Ab + row * 16 + pb + 8);
      unsigned int au[8] = {c0.x, c0.y, c0.z, c0.w, c1.x, c1.y, c1.z, c1.w};
      afrag[mi] = __builtin_bit_cast(v16bf, au);
    }
#pragma unroll
    for (int ni = 0; ni < 2; ++ni) {
      const int col = wn * 32 + ni * 16 + r16;
      const uint4 c0 = *(const uint4*)(Bb + col * 16 + pb);
      const uint4 c1 = *(const uint4*)(Bb + col * 16 + pb + 8);
      unsigned int bu[8] = {c0.x, c0.y, c0.z, c0.w, c1.x, c1.y, c1.z, c1.w};
      bfrag[ni] = __builtin_bit_cast(v16bf, bu);
    }

#pragma unroll
    for (int mi = 0; mi < 4; ++mi)
#pragma unroll
      for (int ni = 0; ni < 2; ++ni)
        acc[mi][ni] = __builtin_amdgcn_wmma_f32_16x16x32_bf16(
            false, afrag[mi], false, bfrag[ni], (short)0, acc[mi][ni], false, false);
  }

  // ---- epilogue: C/D layout (lane 0-15: M=vg, lane 16-31: M=8+vg) + bias fold ----
  const int hi = lane >> 4;
#pragma unroll
  for (int mi = 0; mi < 4; ++mi)
#pragma unroll
    for (int ni = 0; ni < 2; ++ni) {
      const int n = n0 + wn * 32 + ni * 16 + r16;
      const float bias = bi[n] + bh[n];
      const int mbase = m0 + wm * 64 + mi * 16 + hi * 8;
#pragma unroll
      for (int vg = 0; vg < 8; ++vg)
        out[(size_t)(mbase + vg) * G4_D + n] = acc[mi][ni][vg] + bias;
    }
}

// ---------------------------------------------------------------------------
// Persistent BiLSTM recurrence. 16 WGs per direction (grid = 32, block = 128).
// Each WG owns 128 gate rows (GEMV vs transposed bf16 W_hh, L2-resident) and
// 32 cell elements (c kept in registers for all 8192 steps).
// Per-step sync: cluster barrier if launched as a cluster, else atomic spin.
// ---------------------------------------------------------------------------
__device__ __forceinline__ void grid_sync(int* cnt, int target) {
  __syncthreads();
  __threadfence();
  if (threadIdx.x == 0) {
    if (__builtin_amdgcn_cluster_id_x() != 0) {
      __builtin_amdgcn_s_cluster_barrier();          // s_barrier_signal/wait -3
    } else {
      __hip_atomic_fetch_add(cnt, 1, __ATOMIC_ACQ_REL, __HIP_MEMORY_SCOPE_AGENT);
      while (__hip_atomic_load(cnt, __ATOMIC_ACQUIRE, __HIP_MEMORY_SCOPE_AGENT) < target)
        __builtin_amdgcn_s_sleep(1);
    }
  }
  __syncthreads();
}

__global__ __launch_bounds__(128)
void k_lstm_recurrent(const unsigned short* __restrict__ whhT_f,  // [512][2048] bf16
                      const unsigned short* __restrict__ whhT_b,
                      const float* __restrict__ xg,               // [2][S][2048]
                      const float* __restrict__ h0,               // [2][512]
                      const float* __restrict__ c0,               // [2][512]
                      float* __restrict__ h_buf,                  // [2][512]
                      float* __restrict__ gbuf,                   // [2][2048]
                      float* __restrict__ hs,                     // [S][1024]
                      int* __restrict__ cnt) {                    // [2]
  const int wg  = blockIdx.x;
  const int dir = wg >> 4;
  const int w   = wg & 15;
  const int tid = threadIdx.x;

  const unsigned short* WT = dir ? whhT_b : whhT_f;
  const float* xgd = xg + (size_t)dir * S_LEN * G4_D;
  float* hb = h_buf + dir * HD_D;
  float* gb = gbuf + dir * G4_D;
  int* cp = cnt + dir;

  __shared__ float h_s[HD_D];

  const int grow = w * 128 + tid;        // owned gate row (0..2047)
  const int jelem = w * 32 + tid;        // owned cell element (tid < 32)
  float creg = (tid < 32) ? c0[dir * HD_D + jelem] : 0.0f;

  for (int t = 0; t < S_LEN; ++t) {
    const int row_nat = (dir == 0) ? t : (S_LEN - 1 - t);

    // bring h_prev into LDS
    const float* hsrc = (t == 0) ? (h0 + dir * HD_D) : hb;
    for (int k = tid; k < HD_D; k += 128) h_s[k] = hsrc[k];
    __syncthreads();

    // prefetch next step's xg row for this lane (global_prefetch_b8)
    if (t + 1 < S_LEN) {
      const int nrow = (dir == 0) ? (t + 1) : (S_LEN - 2 - t);
      __builtin_prefetch(&xgd[(size_t)nrow * G4_D + grow], 0, 1);
    }

    // phase A: gates[grow] = W_hh[grow,:] . h_prev   (coalesced over lanes)
    float acc = 0.0f;
    const unsigned short* wp = WT + grow;
#pragma unroll 8
    for (int k = 0; k < HD_D; ++k) {
      const unsigned int u = (unsigned int)wp[(size_t)k * G4_D];
      acc = fmaf(bf_lo(u), h_s[k], acc);
    }
    gb[grow] = acc + xgd[(size_t)row_nat * G4_D + grow];   // xg already has b_ih+b_hh

    grid_sync(cp, 16 * (2 * t + 1));

    // phase B: elementwise LSTM cell update on owned elements
    if (tid < 32) {
      const int j = jelem;
      const float gi = gb[j];
      const float gf = gb[HD_D + j];
      const float gg = gb[2 * HD_D + j];
      const float go = gb[3 * HD_D + j];
      creg = sigmoidf_(gf) * creg + sigmoidf_(gi) * tanhf(gg);
      const float h = sigmoidf_(go) * tanhf(creg);
      hb[j] = h;
      hs[(size_t)row_nat * HID_D + dir * HD_D + j] = h;    // bwd writes reversed slot
    }

    grid_sync(cp, 16 * (2 * t + 2));
  }
}

// ---------------------------------------------------------------------------
// feats[s][t] = lstm_out[s] . w_tag[t] + b_tag[t]
// ---------------------------------------------------------------------------
__global__ __launch_bounds__(128)
void k_feats(const float* __restrict__ hs, const float* __restrict__ w_tag,
             const float* __restrict__ b_tag, float* __restrict__ feats) {
  const int s = blockIdx.x;
  __shared__ float row[HID_D];
  __shared__ float part[NT][8];
  const int tid = threadIdx.x;
  for (int k = tid; k < HID_D; k += 128) row[k] = hs[(size_t)s * HID_D + k];
  __syncthreads();
  if (tid < NT * 8) {
    const int t = tid >> 3, seg = tid & 7;
    float a = 0.0f;
    const float* wr = w_tag + t * HID_D + seg * 128;
    const float* rr = row + seg * 128;
#pragma unroll 4
    for (int k = 0; k < 128; ++k) a = fmaf(rr[k], wr[k], a);
    part[t][seg] = a;
  }
  __syncthreads();
  if (tid < NT) {
    float a = b_tag[tid];
    for (int e = 0; e < 8; ++e) a += part[tid][e];
    feats[s * NT + tid] = a;
  }
}

// ---------------------------------------------------------------------------
// CRF: forward logsumexp scan + gold score, single wave32.
// ---------------------------------------------------------------------------
__global__ __launch_bounds__(32)
void k_crf(const float* __restrict__ feats, const float* __restrict__ trans,
           const int* __restrict__ tags, float* __restrict__ out) {
  __shared__ float T[NT][16];
  __shared__ float fv[16];
  __shared__ float fwd_s[1];
  const int lane = threadIdx.x;

  for (int i = lane; i < NT * NT; i += 32) T[i / NT][i % NT] = trans[i];
  if (lane < 16) fv[lane] = (lane == START_TAG) ? 0.0f : NEG_INF;
  __syncthreads();

  for (int s = 0; s < S_LEN; ++s) {
    float nv = 0.0f;
    if (lane < NT) {
      float m = -3.0e38f;
#pragma unroll
      for (int i = 0; i < NT; ++i) m = fmaxf(m, fv[i] + T[lane][i]);
      float sum = 0.0f;
#pragma unroll
      for (int i = 0; i < NT; ++i) sum += __expf(fv[i] + T[lane][i] - m);
      nv = m + __logf(sum) + feats[s * NT + lane];
    }
    __syncthreads();
    if (lane < NT) fv[lane] = nv;
    __syncthreads();
  }

  if (lane == 0) {
    float m = -3.0e38f;
    for (int i = 0; i < NT; ++i) m = fmaxf(m, fv[i] + T[STOP_TAG][i]);
    float sum = 0.0f;
    for (int i = 0; i < NT; ++i) sum += __expf(fv[i] + T[STOP_TAG][i] - m);
    fwd_s[0] = m + __logf(sum);
  }

  // gold score (parallel over lanes, shuffle reduce)
  float g = 0.0f;
  for (int s = lane; s < S_LEN; s += 32) {
    const int cur = tags[s];
    const int prev = (s == 0) ? START_TAG : tags[s - 1];
    g += T[cur][prev] + feats[s * NT + cur];
  }
#pragma unroll
  for (int off = 16; off > 0; off >>= 1) g += __shfl_down(g, off, 32);

  __syncthreads();
  if (lane == 0) {
    g += T[STOP_TAG][tags[S_LEN - 1]];
    out[0] = fwd_s[0] - g;
  }
}

// ---------------------------------------------------------------------------
// Host-side launch
// ---------------------------------------------------------------------------
extern "C" void kernel_launch(void* const* d_in, const int* in_sizes, int n_in,
                              void* d_out, int out_size, void* d_ws, size_t ws_size,
                              hipStream_t stream) {
  (void)in_sizes; (void)n_in; (void)out_size; (void)ws_size;
  const float* embeds = (const float*)d_in[0];
  const int*   tags   = (const int*)d_in[1];
  const float* w_ih_f = (const float*)d_in[2];
  const float* w_hh_f = (const float*)d_in[3];
  const float* b_ih_f = (const float*)d_in[4];
  const float* b_hh_f = (const float*)d_in[5];
  const float* w_ih_b = (const float*)d_in[6];
  const float* w_hh_b = (const float*)d_in[7];
  const float* b_ih_b = (const float*)d_in[8];
  const float* b_hh_b = (const float*)d_in[9];
  const float* w_tag  = (const float*)d_in[10];
  const float* b_tag  = (const float*)d_in[11];
  const float* trans  = (const float*)d_in[12];
  const float* h0     = (const float*)d_in[13];
  const float* c0     = (const float*)d_in[14];
  float* out = (float*)d_out;

  char* p = (char*)d_ws;
  auto alloc = [&](size_t bytes) -> char* {
    char* r = p;
    p += (bytes + 255) & ~(size_t)255;
    return r;
  };
  unsigned short* Xb    = (unsigned short*)alloc((size_t)S_LEN * EMB_D * 2);
  unsigned short* Wihf  = (unsigned short*)alloc((size_t)G4_D * EMB_D * 2);
  unsigned short* Wihb  = (unsigned short*)alloc((size_t)G4_D * EMB_D * 2);
  unsigned short* WhTf  = (unsigned short*)alloc((size_t)HD_D * G4_D * 2);
  unsigned short* WhTb  = (unsigned short*)alloc((size_t)HD_D * G4_D * 2);
  float* xg    = (float*)alloc((size_t)2 * S_LEN * G4_D * 4);
  float* hs    = (float*)alloc((size_t)S_LEN * HID_D * 4);
  float* hbuf  = (float*)alloc((size_t)2 * HD_D * 4);
  float* gbuf  = (float*)alloc((size_t)2 * G4_D * 4);
  float* feats = (float*)alloc((size_t)S_LEN * NT * 4);
  int*   cnt   = (int*)alloc(256);

  k_init_sync<<<1, 32, 0, stream>>>(cnt);

  {
    int n = S_LEN * EMB_D;
    k_f32_to_bf16<<<(n + 255) / 256, 256, 0, stream>>>(embeds, Xb, n);
    n = G4_D * EMB_D;
    k_f32_to_bf16<<<(n + 255) / 256, 256, 0, stream>>>(w_ih_f, Wihf, n);
    k_f32_to_bf16<<<(n + 255) / 256, 256, 0, stream>>>(w_ih_b, Wihb, n);
    n = G4_D * HD_D;
    k_whh_transpose<<<(n + 255) / 256, 256, 0, stream>>>(w_hh_f, WhTf);
    k_whh_transpose<<<(n + 255) / 256, 256, 0, stream>>>(w_hh_b, WhTb);
  }

  {
    dim3 grid(G4_D / BN, S_LEN / BM, 2);
    k_xproj_wmma<<<grid, 256, 0, stream>>>(Xb, Wihf, Wihb,
                                           b_ih_f, b_hh_f, b_ih_b, b_hh_b, xg);
  }

  k_lstm_recurrent<<<32, 128, 0, stream>>>(WhTf, WhTb, xg, h0, c0,
                                           hbuf, gbuf, hs, cnt);

  k_feats<<<S_LEN, 128, 0, stream>>>(hs, w_tag, b_tag, feats);

  k_crf<<<1, 32, 0, stream>>>(feats, trans, tags, out);
}
